// GraphNet_80590766342779
// MI455X (gfx1250) — matrix operand order repaired and checked by reading, not compiled, not run on hardware
//
#include <hip/hip_runtime.h>
#include <hip/hip_bf16.h>

// GraphNet message passing for MI455X (gfx1250, wave32, WMMA).
// Edge MLP restructured: layer1 = Y[b,j] + C[b,i] + wd*dist (precomputed GEMMs),
// layers 2/3 = v_wmma_f32_16x16x32_f16 tiles, sum-over-j via ds_add_f32.

typedef _Float16 v16h __attribute__((ext_vector_type(16)));
typedef _Float16 v8h  __attribute__((ext_vector_type(8)));
typedef float    v8f  __attribute__((ext_vector_type(8)));

#define NB   8
#define NN   256
#define HID  32
#define F1   64   // edge layer1/2 width
#define F3   32   // edge layer3 width

__device__ __forceinline__ float lrelu(float v) { return v >= 0.f ? v : 0.1f * v; }

__device__ __forceinline__ v8f wmma16(v16h a, v16h b, v8f c) {
    return __builtin_amdgcn_wmma_f32_16x16x32_f16(
        /*neg_a=*/false, a, /*neg_b=*/false, b,
        /*c_mod=*/(short)0, c, /*reuse_a=*/false, /*reuse_b=*/false);
}

// A-fragment (16x32 f16, ISA 7.12.2): lane row = l&15, khalf = l>>4:
// VGPR0-3 hold K = khalf*8 .. +7, VGPR4-7 hold K = 16+khalf*8 .. +7.
__device__ __forceinline__ v16h load_frag_A(const _Float16* buf, int rowBase,
                                            int stride, int kBase, int lane) {
    int row = rowBase + (lane & 15);
    int kh  = (lane >> 4) * 8;
    const _Float16* p = buf + row * stride + kBase + kh;
    v8h lo = *(const v8h*)p;
    v8h hi = *(const v8h*)(p + 16);
    return __builtin_shufflevector(lo, hi, 0,1,2,3,4,5,6,7,8,9,10,11,12,13,14,15);
}

// B-fragment = W^T (32x16): lane col = l&15, khalf = l>>4:
// VGPR0-3 hold K = khalf*8 .. +7, VGPR4-7 hold K = 16+khalf*8 .. +7.
__device__ __forceinline__ v16h load_frag_B(const _Float16* w, int oBase,
                                            int stride, int kBase, int lane) {
    int o = oBase + (lane & 15);
    int k = kBase + (lane >> 4) * 8;
    const _Float16* p = w + o * stride + k;
    v8h lo = *(const v8h*)p;
    v8h hi = *(const v8h*)(p + 16);
    return __builtin_shufflevector(lo, hi, 0,1,2,3,4,5,6,7,8,9,10,11,12,13,14,15);
}

// ---------------- pad x (B,N,4) -> X (B,N,32) ----------------
__global__ __launch_bounds__(256) void gn_pad_kernel(const float* __restrict__ xin,
                                                     float* __restrict__ X) {
    int idx = blockIdx.x * 256 + threadIdx.x;           // over B*N*32
    if (idx >= NB * NN * HID) return;
    int k  = idx & (HID - 1);
    int bn = idx >> 5;
    X[idx] = (k < 4) ? xin[bn * 4 + k] : 0.f;
}

// ------------- per-batch precompute: Y = X@Wb^T (f16), C = X@Wa^T + b1 (f32) -------------
__global__ __launch_bounds__(256) void gn_precompute_kernel(
    const float* __restrict__ X, const float* __restrict__ W1,
    const float* __restrict__ b1, _Float16* __restrict__ Y, float* __restrict__ C) {
    __shared__ __align__(16) _Float16 sX[NN * HID];     // 16KB
    __shared__ __align__(16) _Float16 sWa[F1 * HID];    // 4KB
    __shared__ __align__(16) _Float16 sWb[F1 * HID];    // 4KB
    __shared__ float sB1[F1];

    const int t = threadIdx.x;
    const int b = blockIdx.x;
    const float* Xb = X + (size_t)b * NN * HID;
    for (int idx = t; idx < NN * HID; idx += 256) sX[idx] = (_Float16)Xb[idx];
    for (int idx = t; idx < F1 * HID; idx += 256) {
        int o = idx >> 5, k = idx & 31;
        sWa[idx] = (_Float16)W1[o * 65 + k];            // xi part
        sWb[idx] = (_Float16)W1[o * 65 + 32 + k];       // xj part
    }
    if (t < F1) sB1[t] = b1[t];
    __syncthreads();

    const int lane = t & 31, wave = t >> 5;
    _Float16* Yb = Y + (size_t)b * NN * F1;
    float*    Cb = C + (size_t)b * NN * F1;
    #pragma unroll
    for (int u = 0; u < 2; ++u) {
        int mt = wave * 2 + u;
        v16h a = load_frag_A(sX, mt * 16, HID, 0, lane);
        #pragma unroll
        for (int nt = 0; nt < 4; ++nt) {
            int col   = nt * 16 + (lane & 15);
            int rbase = mt * 16 + (lane >> 4) * 8;
            v8f accY = {};
            accY = wmma16(a, load_frag_B(sWb, nt * 16, HID, 0, lane), accY);
            #pragma unroll
            for (int v = 0; v < 8; ++v)
                Yb[(size_t)(rbase + v) * F1 + col] = (_Float16)accY[v];
            v8f accC = {};
            accC = wmma16(a, load_frag_B(sWa, nt * 16, HID, 0, lane), accC);
            float bias = sB1[col];
            #pragma unroll
            for (int v = 0; v < 8; ++v)
                Cb[(size_t)(rbase + v) * F1 + col] = accC[v] + bias;
        }
    }
}

// ------------- edge MLP (layers 1-3) + sum over j + fused node MLP; one block per (b,i) -------------
__global__ __launch_bounds__(256) void gn_edge_node_kernel(
    const float* __restrict__ X, const _Float16* __restrict__ Y,
    const float* __restrict__ C, const float* __restrict__ W1,
    const float* __restrict__ W2f, const float* __restrict__ b2,
    const float* __restrict__ W3f, const float* __restrict__ b3,
    const float* __restrict__ Wn1, const float* __restrict__ bn1,
    const float* __restrict__ Wn2, const float* __restrict__ bn2,
    const float* __restrict__ Wn3, const float* __restrict__ bn3,
    float* __restrict__ Xout, int outDim) {
    __shared__ __align__(16) _Float16 sH1[NN * F1];       // 32KB: Y staged, then H1 in place
    __shared__ __align__(16) _Float16 sW2[F1 * F1];       // 8KB
    __shared__ __align__(16) _Float16 sW3[F3 * F1];       // 4KB
    __shared__ __align__(16) _Float16 sH2[8][16 * F1];    // 16KB per-wave scratch
    __shared__ float sC[F1], sWd[F1], sB2[F1], sB3[F3];
    __shared__ float sXi[HID], sDist[NN], sEsum[F3];
    __shared__ float sXc[2 * HID], sN1[HID], sN2[HID];

    const int t  = threadIdx.x;
    const int bi = blockIdx.x;          // b*N + i
    const int b  = bi >> 8;

    // Stage Y[b] -> LDS (uint4 copies), convert weights f32->f16, load vectors.
    {
        const uint4* src = (const uint4*)(Y + (size_t)b * NN * F1);
        uint4* dst = (uint4*)sH1;
        #pragma unroll
        for (int idx = t; idx < NN * F1 * 2 / 16; idx += 256) dst[idx] = src[idx];
    }
    for (int idx = t; idx < F1 * F1; idx += 256) sW2[idx] = (_Float16)W2f[idx];
    for (int idx = t; idx < F3 * F1; idx += 256) sW3[idx] = (_Float16)W3f[idx];
    if (t < F1) { sC[t] = C[(size_t)bi * F1 + t]; sWd[t] = W1[t * 65 + 64]; sB2[t] = b2[t]; }
    if (t < F3) { sB3[t] = b3[t]; sEsum[t] = 0.f; }
    if (t < HID) sXi[t] = X[(size_t)bi * HID + t];
    __syncthreads();

    // dist[j] in f32 (j = t); then H1 row j in place: lrelu(Y + C_i + wd*dist).
    {
        float acc = 0.f;
        const float* xj = X + ((size_t)b * NN + t) * HID;
        #pragma unroll
        for (int k = 0; k < HID; ++k) { float d = xj[k] - sXi[k] + 1e-12f; acc += d * d; }
        sDist[t] = acc;
        float dj = acc;
        _Float16* row = sH1 + t * F1;
        #pragma unroll
        for (int o = 0; o < F1; ++o) {
            float v = (float)row[o] + sC[o] + sWd[o] * dj;
            row[o] = (_Float16)lrelu(v);
        }
    }
    __syncthreads();

    const int lane = t & 31, wave = t >> 5;
    _Float16* scr = sH2[wave];
    #pragma unroll
    for (int u = 0; u < 2; ++u) {
        int mt = wave * 2 + u;
        // ---- layer 2: H2 = lrelu(H1 @ W2^T + b2), 16x64 tile ----
        v16h a0 = load_frag_A(sH1, mt * 16, F1, 0, lane);
        v16h a1 = load_frag_A(sH1, mt * 16, F1, 32, lane);
        #pragma unroll
        for (int nt = 0; nt < 4; ++nt) {
            v8f acc = {};
            acc = wmma16(a0, load_frag_B(sW2, nt * 16, F1, 0, lane), acc);
            acc = wmma16(a1, load_frag_B(sW2, nt * 16, F1, 32, lane), acc);
            int col  = nt * 16 + (lane & 15);
            int rloc = (lane >> 4) * 8;
            float bias = sB2[col];
            #pragma unroll
            for (int v = 0; v < 8; ++v)
                scr[(rloc + v) * F1 + col] = (_Float16)lrelu(acc[v] + bias);
        }
        // Same-wave LDS ops are in-order (DScnt) -> no barrier needed.
        // ---- layer 3: H3 = lrelu(H2 @ W3^T + b3); reduce over rows j ----
        v16h c0 = load_frag_A(scr, 0, F1, 0, lane);
        v16h c1 = load_frag_A(scr, 0, F1, 32, lane);
        #pragma unroll
        for (int nt = 0; nt < 2; ++nt) {
            v8f acc = {};
            acc = wmma16(c0, load_frag_B(sW3, nt * 16, F1, 0, lane), acc);
            acc = wmma16(c1, load_frag_B(sW3, nt * 16, F1, 32, lane), acc);
            int col = nt * 16 + (lane & 15);
            float bias = sB3[col];
            float part = 0.f;
            #pragma unroll
            for (int v = 0; v < 8; ++v) part += lrelu(acc[v] + bias);
            atomicAdd(&sEsum[col], part);       // ds_add_f32
        }
    }
    __syncthreads();

    // ---- fused node MLP on [e | x_i] (tiny; one wave does the math) ----
    if (t < HID) { sXc[t] = sEsum[t]; sXc[HID + t] = sXi[t]; }
    __syncthreads();
    if (t < HID) {
        float acc = bn1[t];
        #pragma unroll
        for (int k = 0; k < 2 * HID; ++k) acc += Wn1[t * (2 * HID) + k] * sXc[k];
        sN1[t] = lrelu(acc);
    }
    __syncthreads();
    if (t < HID) {
        float acc = bn2[t];
        #pragma unroll
        for (int k = 0; k < HID; ++k) acc += Wn2[t * HID + k] * sN1[k];
        sN2[t] = lrelu(acc);
    }
    __syncthreads();
    if (t < outDim) {
        float acc = bn3[t];
        #pragma unroll
        for (int k = 0; k < HID; ++k) acc += Wn3[t * HID + k] * sN2[k];
        Xout[(size_t)bi * outDim + t] = lrelu(acc);
    }
}

extern "C" void kernel_launch(void* const* d_in, const int* in_sizes, int n_in,
                              void* d_out, int out_size, void* d_ws, size_t ws_size,
                              hipStream_t stream) {
    (void)in_sizes; (void)n_in; (void)out_size; (void)ws_size;
    const float* x = (const float*)d_in[0];

    // Workspace layout (needs 1.25 MB):
    char* ws = (char*)d_ws;
    float*    X0 = (float*)ws;                        // (B,N,32) f32  256KB
    float*    X1 = (float*)(ws + 262144);             // (B,N,32) f32  256KB
    _Float16* Y  = (_Float16*)(ws + 524288);          // (B,N,64) f16  256KB
    float*    Cm = (float*)(ws + 786432);             // (B,N,64) f32  512KB

    gn_pad_kernel<<<(NB * NN * HID + 255) / 256, 256, 0, stream>>>(x, X0);

    for (int r = 0; r < 2; ++r) {
        const float* W1  = (const float*)d_in[1 + 6 * r];
        const float* b1  = (const float*)d_in[2 + 6 * r];
        const float* W2  = (const float*)d_in[3 + 6 * r];
        const float* b2  = (const float*)d_in[4 + 6 * r];
        const float* W3  = (const float*)d_in[5 + 6 * r];
        const float* b3  = (const float*)d_in[6 + 6 * r];
        const float* Wn1 = (const float*)d_in[13 + 6 * r];
        const float* bn1 = (const float*)d_in[14 + 6 * r];
        const float* Wn2 = (const float*)d_in[15 + 6 * r];
        const float* bn2 = (const float*)d_in[16 + 6 * r];
        const float* Wn3 = (const float*)d_in[17 + 6 * r];
        const float* bn3 = (const float*)d_in[18 + 6 * r];

        const float* Xcur = (r == 0) ? X0 : X1;
        float* Xout = (r == 0) ? X1 : (float*)d_out;
        int outDim  = (r == 0) ? HID : 4;

        gn_precompute_kernel<<<NB, 256, 0, stream>>>(Xcur, W1, b1, Y, Cm);
        gn_edge_node_kernel<<<NB * NN, 256, 0, stream>>>(
            Xcur, Y, Cm, W1, W2, b2, W3, b3,
            Wn1, bn1, Wn2, bn2, Wn3, bn3, Xout, outDim);
    }
}
